// CustomGraphConv_57561151701651
// MI455X (gfx1250) — matrix operand order repaired and checked by reading, not compiled, not run on hardware
//
#include <hip/hip_runtime.h>

#define NN 8192
#define FI 256
#define FO 256

typedef __attribute__((ext_vector_type(16))) __bf16 v16bf;
typedef __attribute__((ext_vector_type(8)))  __bf16 v8bf;
typedef __attribute__((ext_vector_type(8)))  float  v8f;
typedef __attribute__((ext_vector_type(4)))  float  v4f;

// ---- WMMA fragment loaders (layouts per CDNA5 ISA §7.12.2) ----

// A-matrix 16x32 bf16, row-major source [rows][ldk].
// lane<16: M=lane, elems 0..7 = K kk+0..7, elems 8..15 = K kk+16..23
// lane>=16: M=lane-16, elems 0..7 = K kk+8..15, elems 8..15 = K kk+24..31
static __device__ __forceinline__ v16bf ld_afrag_bf(const __bf16* __restrict__ base,
                                                    int ldk, int r0, int kk, int lane) {
  const __bf16* q = base + (size_t)(r0 + (lane & 15)) * ldk + kk + ((lane >> 4) << 3);
  v8bf a = *(const v8bf*)q;
  v8bf b = *(const v8bf*)(q + 16);
  v16bf r;
#pragma unroll
  for (int i = 0; i < 8; ++i) { r[i] = a[i]; r[i + 8] = b[i]; }
  return r;
}

// B-matrix 32x16 bf16 from column-major (transposed) storage [cols][ldk].
// lane<16: N=lane, elems 0..7 = K kk+0..7, elems 8..15 = K kk+16..23 ; lane>=16: N=lane-16, K kk+8..15, kk+24..31
static __device__ __forceinline__ v16bf ld_bfragT(const __bf16* __restrict__ base,
                                                  int ldk, int c0, int kk, int lane) {
  const __bf16* q = base + (size_t)(c0 + (lane & 15)) * ldk + kk + ((lane >> 4) << 3);
  v8bf a = *(const v8bf*)q;
  v8bf b = *(const v8bf*)(q + 16);
  v16bf r;
#pragma unroll
  for (int i = 0; i < 8; ++i) { r[i] = a[i]; r[i + 8] = b[i]; }
  return r;
}

// ---- Kernel 1: transpose + hi/lo split of W (256x256) ----
__global__ __launch_bounds__(256) void splitwt_kernel(const float* __restrict__ W,
                                                      __bf16* __restrict__ Wth,
                                                      __bf16* __restrict__ Wtl) {
  int i = blockIdx.x * 256 + threadIdx.x;      // i = r*256 + c  (W[r][c], r=K, c=out-feature)
  int r = i >> 8, c = i & 255;
  float x = W[i];
  __bf16 h = (__bf16)x;
  float rem = x - (float)h;
  Wth[(size_t)c * FI + r] = h;
  Wtl[(size_t)c * FI + r] = (__bf16)rem;
}

// ---- Kernel 2: hi/lo split of X (8192x256, row-major) ----
__global__ __launch_bounds__(256) void splitx_kernel(const float* __restrict__ X,
                                                     __bf16* __restrict__ Xh,
                                                     __bf16* __restrict__ Xl) {
  int i = blockIdx.x * 256 + threadIdx.x;
  float x = X[i];
  __bf16 h = (__bf16)x;
  Xh[i] = h;
  Xl[i] = (__bf16)(x - (float)h);
}

// ---- Kernel 3: d[i] = (1 + rowsum(A))^-1/2 ; sums of 0/1 are exact in fp32 ----
__global__ __launch_bounds__(256) void degree_kernel(const float* __restrict__ A,
                                                     float* __restrict__ dvec) {
  __shared__ float red[256];
  int row = blockIdx.x;
  const float* p = A + (size_t)row * NN;
  float s = 0.f;
  for (int j = threadIdx.x; j < NN; j += 256) s += p[j];
  red[threadIdx.x] = s;
  __syncthreads();
  for (int off = 128; off > 0; off >>= 1) {
    if (threadIdx.x < off) red[threadIdx.x] += red[threadIdx.x + off];
    __syncthreads();
  }
  if (threadIdx.x == 0) dvec[row] = (float)(1.0 / sqrt((double)(1.0f + red[0])));
}

// ---- Kernel 4: H = X@W via 3-term bf16x2 WMMA; G = d ⊙ H split hi/lo, column-major ----
__global__ __launch_bounds__(256) void hgemm_kernel(const __bf16* __restrict__ Xh,
                                                    const __bf16* __restrict__ Xl,
                                                    const __bf16* __restrict__ Wth,
                                                    const __bf16* __restrict__ Wtl,
                                                    const float* __restrict__ dvec,
                                                    __bf16* __restrict__ Gh,
                                                    __bf16* __restrict__ Gl) {
  int lane = threadIdx.x & 31;
  int wave = threadIdx.x >> 5;
  int tile = blockIdx.x * 8 + wave;            // 8192 tiles = 512 row-tiles x 16 col-tiles
  int r0 = (tile >> 4) << 4;
  int c0 = (tile & 15) << 4;
  v8f acc = {0.f, 0.f, 0.f, 0.f, 0.f, 0.f, 0.f, 0.f};
#pragma unroll 2
  for (int kk = 0; kk < FI; kk += 32) {
    v16bf ah = ld_afrag_bf(Xh, FI, r0, kk, lane);
    v16bf al = ld_afrag_bf(Xl, FI, r0, kk, lane);
    v16bf bh = ld_bfragT(Wth, FI, c0, kk, lane);
    v16bf bl = ld_bfragT(Wtl, FI, c0, kk, lane);
    acc = __builtin_amdgcn_wmma_f32_16x16x32_bf16(false, ah, false, bh, (short)0, acc, false, false);
    acc = __builtin_amdgcn_wmma_f32_16x16x32_bf16(false, ah, false, bl, (short)0, acc, false, false);
    acc = __builtin_amdgcn_wmma_f32_16x16x32_bf16(false, al, false, bh, (short)0, acc, false, false);
  }
  // C/D layout: lane<16 -> rows r0+v, col c0+lane ; lane>=16 -> rows r0+8+v
  int col = c0 + (lane & 15);
  int rbase = r0 + ((lane >> 4) << 3);
#pragma unroll
  for (int v = 0; v < 8; ++v) {
    int row = rbase + v;
    float g = acc[v] * dvec[row];               // G = d_k * H[k][c]
    __bf16 gh = (__bf16)g;
    float rem = g - (float)gh;
    Gh[(size_t)col * NN + row] = gh;            // column-major for contiguous B-frags
    Gl[(size_t)col * NN + row] = (__bf16)rem;
  }
}

// ---- Kernel 5: out = d ⊙ ((I+A) @ (Ghi+Glo)) + bias ----
// Block = 8 waves, 16 rows x 256 cols; wave w owns cols [32w, 32w+32) (two 16x16 tiles).
__global__ __launch_bounds__(256) void agemm_kernel(const float* __restrict__ A,
                                                    const __bf16* __restrict__ Gh,
                                                    const __bf16* __restrict__ Gl,
                                                    const float* __restrict__ dvec,
                                                    const float* __restrict__ bias,
                                                    float* __restrict__ out) {
  int lane = threadIdx.x & 31;
  int wave = threadIdx.x >> 5;
  int r0 = blockIdx.x << 4;
  int c0 = wave << 5;
  int row = r0 + (lane & 15);
  int seg = (lane >> 4) << 3;                   // 0 or 8
  v8f acc0 = {0.f, 0.f, 0.f, 0.f, 0.f, 0.f, 0.f, 0.f};
  v8f acc1 = {0.f, 0.f, 0.f, 0.f, 0.f, 0.f, 0.f, 0.f};

#pragma unroll 2
  for (int kk = 0; kk < NN; kk += 32) {
    // Build A-fragment in-register from fp32 A (0/1 values -> exact in bf16), add I.
    int cbase = kk + seg;
    const float* q = A + (size_t)row * NN + cbase;
    v4f x0 = *(const v4f*)q;
    v4f x1 = *(const v4f*)(q + 4);
    v4f x2 = *(const v4f*)(q + 16);
    v4f x3 = *(const v4f*)(q + 20);
    float f[16];
#pragma unroll
    for (int i = 0; i < 4; ++i) { f[i] = x0[i]; f[4 + i] = x1[i]; f[8 + i] = x2[i]; f[12 + i] = x3[i]; }
    if ((kk >> 5) == (r0 >> 5)) {               // uniform branch: diagonal lives in this K-block
#pragma unroll
      for (int i = 0; i < 16; ++i) {
        int colg = (i < 8) ? (cbase + i) : (cbase + 8 + i);  // elems 8..15 hold K+16..23
        f[i] += (colg == row) ? 1.0f : 0.0f;
      }
    }
    v16bf a;
#pragma unroll
    for (int i = 0; i < 16; ++i) a[i] = (__bf16)f[i];

    v16bf b0h = ld_bfragT(Gh, NN, c0, kk, lane);
    v16bf b0l = ld_bfragT(Gl, NN, c0, kk, lane);
    v16bf b1h = ld_bfragT(Gh, NN, c0 + 16, kk, lane);
    v16bf b1l = ld_bfragT(Gl, NN, c0 + 16, kk, lane);

    acc0 = __builtin_amdgcn_wmma_f32_16x16x32_bf16(false, a, false, b0h, (short)0, acc0, false, false);
    acc0 = __builtin_amdgcn_wmma_f32_16x16x32_bf16(false, a, false, b0l, (short)0, acc0, false, false);
    acc1 = __builtin_amdgcn_wmma_f32_16x16x32_bf16(false, a, false, b1h, (short)0, acc1, false, false);
    acc1 = __builtin_amdgcn_wmma_f32_16x16x32_bf16(false, a, false, b1l, (short)0, acc1, false, false);
  }

  // Epilogue: out[r][c] = d_r * acc + bias[c]
  int n = lane & 15;
  int rb = r0 + ((lane >> 4) << 3);
  float b0 = bias[c0 + n];
  float b1 = bias[c0 + 16 + n];
#pragma unroll
  for (int v = 0; v < 8; ++v) {
    int rr = rb + v;
    float di = dvec[rr];
    out[(size_t)rr * FO + c0 + n]      = di * acc0[v] + b0;
    out[(size_t)rr * FO + c0 + 16 + n] = di * acc1[v] + b1;
  }
}

extern "C" void kernel_launch(void* const* d_in, const int* in_sizes, int n_in,
                              void* d_out, int out_size, void* d_ws, size_t ws_size,
                              hipStream_t stream) {
  (void)in_sizes; (void)n_in; (void)out_size; (void)ws_size;
  const float* A    = (const float*)d_in[0];   // [8192,8192]
  const float* X    = (const float*)d_in[1];   // [8192,256]
  const float* W    = (const float*)d_in[2];   // [256,256]
  const float* bias = (const float*)d_in[3];   // [256]
  float* out = (float*)d_out;                  // [8192,256]

  // Workspace layout (~16.3 MB total)
  char* ws = (char*)d_ws;
  const size_t SZ_D  = (size_t)NN * sizeof(float);            // 32 KB
  const size_t SZ_X  = (size_t)NN * FI * sizeof(__bf16);      // 4 MB each
  const size_t SZ_W  = (size_t)FI * FO * sizeof(__bf16);      // 128 KB each
  const size_t SZ_G  = (size_t)NN * FO * sizeof(__bf16);      // 4 MB each
  float*  dvec = (float*) (ws);
  __bf16* Xh   = (__bf16*)(ws + SZ_D);
  __bf16* Xl   = (__bf16*)(ws + SZ_D + SZ_X);
  __bf16* Wth  = (__bf16*)(ws + SZ_D + 2 * SZ_X);
  __bf16* Wtl  = (__bf16*)(ws + SZ_D + 2 * SZ_X + SZ_W);
  __bf16* Gh   = (__bf16*)(ws + SZ_D + 2 * SZ_X + 2 * SZ_W);
  __bf16* Gl   = (__bf16*)(ws + SZ_D + 2 * SZ_X + 2 * SZ_W + SZ_G);

  splitwt_kernel<<<(FI * FO) / 256, 256, 0, stream>>>(W, Wth, Wtl);
  splitx_kernel<<<(NN * FI) / 256, 256, 0, stream>>>(X, Xh, Xl);
  degree_kernel<<<NN, 256, 0, stream>>>(A, dvec);
  hgemm_kernel<<<(NN / 16) * (FO / 16) / 8, 256, 0, stream>>>(Xh, Xl, Wth, Wtl, dvec, Gh, Gl);
  agemm_kernel<<<NN / 16, 256, 0, stream>>>(A, Gh, Gl, dvec, bias, out);
}
